// EdgeMessagePassingNet_36197984370736
// MI455X (gfx1250) — matrix operand order repaired and checked by reading, not compiled, not run on hardware
//
#include <hip/hip_runtime.h>
#include <hip/hip_bf16.h>

#define N_NODES 50000
#define N_EDGES 800000
#define IN_C    128
#define HID     64
#define X_CAT_C 192            // HID * 3 layers, concatenated
#define EDGE_K  384            // 2 * X_CAT_C
#define EK_STRIDE 392          // padded LDS row stride (bf16 elems, mult of 8)

typedef __attribute__((ext_vector_type(16))) __bf16 v16bf;
typedef __attribute__((ext_vector_type(8)))  __bf16 v8bf;
typedef __attribute__((ext_vector_type(8)))  float  v8f;

// 16-byte vector load of 8 contiguous bf16 (works for global and LDS pointers)
__device__ __forceinline__ v8bf ld8(const __bf16* p) {
    union { uint4 u; v8bf v; } t;
    t.u = *(const uint4*)p;
    return t.v;
}
// A fragment (16x32 bf16): elems 0..7 -> K = kb+hh*8+i, elems 8..15 -> K = kb+16+hh*8+i
__device__ __forceinline__ v16bf load_afrag(const __bf16* row, int kb, int hh) {
    v8bf lo = ld8(row + kb + hh * 8);
    v8bf hi = ld8(row + kb + 16 + hh * 8);
    return __builtin_shufflevector(lo, hi, 0,1,2,3,4,5,6,7,8,9,10,11,12,13,14,15);
}
// B fragment (32x16 bf16): elems 0..7 -> K = kb+hh*8+i, elems 8..15 -> K = kb+16+hh*8+i, col = lane&15
__device__ __forceinline__ v16bf load_bfrag(const __bf16* row, int kb, int hh) {
    const __bf16* p = row + kb + hh * 8;
    v8bf lo = ld8(p);
    v8bf hi = ld8(p + 16);
    return __builtin_shufflevector(lo, hi, 0,1,2,3,4,5,6,7,8,9,10,11,12,13,14,15);
}

// ---------------------------------------------------------------------------
__global__ void k_degree(const int* __restrict__ dst, float* __restrict__ deg) {
    int e = blockIdx.x * blockDim.x + threadIdx.x;
    if (e < N_EDGES) atomicAdd(&deg[dst[e]], 1.0f);
}

// msg[dst[e], f] += h[src[e], f]   (one block per edge, fin threads)
__global__ void k_scatter(const float* __restrict__ h, int hstride, int fin,
                          const int* __restrict__ src, const int* __restrict__ dst,
                          float* __restrict__ msg) {
    int e = blockIdx.x;
    int f = threadIdx.x;
    int s = src[e], d = dst[e];
    atomicAdd(&msg[(size_t)d * fin + f], h[(size_t)s * hstride + f]);
}

// Build node-layer A = [msg/deg | h] as bf16 hi/lo.  grid = N, block = 2*fin
__global__ void k_prep_node_A(const float* __restrict__ msg,
                              const float* __restrict__ deg,
                              const float* __restrict__ hprev, int hstride, int fin,
                              __bf16* __restrict__ Ah, __bf16* __restrict__ Al) {
    int n = blockIdx.x;
    int k = threadIdx.x;
    float inv = 1.0f / fmaxf(deg[n], 1.0f);
    float v = (k < fin) ? msg[(size_t)n * fin + k] * inv
                        : hprev[(size_t)n * hstride + (k - fin)];
    __bf16 h = (__bf16)v;
    size_t o = (size_t)n * (2 * fin) + k;
    Ah[o] = h;
    Al[o] = (__bf16)(v - (float)h);
}

// Build per-layer B = [Wl | Wr] as bf16 hi/lo.  grid = 64, block = 2*fin
__global__ void k_prep_wcat(const float* __restrict__ Wl, const float* __restrict__ Wr,
                            int fin, __bf16* __restrict__ Wh, __bf16* __restrict__ Wlo) {
    int o = blockIdx.x, k = threadIdx.x;
    float v = (k < fin) ? Wl[(size_t)o * fin + k] : Wr[(size_t)o * fin + (k - fin)];
    __bf16 h = (__bf16)v;
    size_t idx = (size_t)o * (2 * fin) + k;
    Wh[idx] = h;
    Wlo[idx] = (__bf16)(v - (float)h);
}

// Wm1 [64, 384] -> bf16 hi/lo.  grid = 64, block = 384
__global__ void k_prep_wm1(const float* __restrict__ Wm1,
                           __bf16* __restrict__ Wh, __bf16* __restrict__ Wlo) {
    size_t idx = (size_t)blockIdx.x * EDGE_K + threadIdx.x;
    float v = Wm1[idx];
    __bf16 h = (__bf16)v;
    Wh[idx] = h;
    Wlo[idx] = (__bf16)(v - (float)h);
}

// ---------------------------------------------------------------------------
// SAGE GEMM: relu(A @ B.T + bias) -> xcat slice (f32 + bf16 hi/lo)
// Wave tile = 16 nodes x 16 outputs; 4 waves/block = 64 outputs. bf16x3.
// ---------------------------------------------------------------------------
template <int KTOT>
__global__ void k_sage_gemm(const __bf16* __restrict__ Ah, const __bf16* __restrict__ Al,
                            const __bf16* __restrict__ Bh, const __bf16* __restrict__ Bl,
                            const float* __restrict__ bias,
                            float*  __restrict__ xcat,
                            __bf16* __restrict__ xcat_h, __bf16* __restrict__ xcat_l) {
    const int lane = threadIdx.x & 31;
    const int wave = threadIdx.x >> 5;
    const int m    = lane & 15;
    const int hh   = lane >> 4;
    const int ocol = wave * 16 + m;

    const __bf16* ah_row = Ah + (size_t)(blockIdx.x * 16 + m) * KTOT;
    const __bf16* al_row = Al + (size_t)(blockIdx.x * 16 + m) * KTOT;
    const __bf16* bh_row = Bh + (size_t)ocol * KTOT;
    const __bf16* bl_row = Bl + (size_t)ocol * KTOT;

    v8f c = {};
    #pragma unroll
    for (int kb = 0; kb < KTOT; kb += 32) {
        v16bf a_hi = load_afrag(ah_row, kb, hh);
        v16bf a_lo = load_afrag(al_row, kb, hh);
        v16bf b_hi = load_bfrag(bh_row, kb, hh);
        v16bf b_lo = load_bfrag(bl_row, kb, hh);
        c = __builtin_amdgcn_wmma_f32_16x16x32_bf16(false, a_hi, false, b_hi, (short)0, c, false, false);
        c = __builtin_amdgcn_wmma_f32_16x16x32_bf16(false, a_hi, false, b_lo, (short)0, c, false, false);
        c = __builtin_amdgcn_wmma_f32_16x16x32_bf16(false, a_lo, false, b_hi, (short)0, c, false, false);
    }

    const float b = bias[ocol];
    #pragma unroll
    for (int r = 0; r < 8; ++r) {
        int n2 = blockIdx.x * 16 + r + hh * 8;
        float v = fmaxf(c[r] + b, 0.0f);
        size_t o = (size_t)n2 * X_CAT_C + ocol;
        xcat[o] = v;
        __bf16 vh = (__bf16)v;
        xcat_h[o] = vh;
        xcat_l[o] = (__bf16)(v - (float)vh);
    }
}

// ---------------------------------------------------------------------------
// Edge MLP: out[e] = relu(concat(xcat[src],xcat[dst]) @ Wm1.T + bm1) @ Wm2.T + bm2
// Block = 128 thr (4 waves) = 16 edges; LDS-staged bf16 hi/lo tiles.
// ---------------------------------------------------------------------------
__global__ void k_edge_mlp(const __bf16* __restrict__ xcat_h, const __bf16* __restrict__ xcat_l,
                           const int* __restrict__ src, const int* __restrict__ dst,
                           const __bf16* __restrict__ Wh, const __bf16* __restrict__ Wlo,
                           const float* __restrict__ bm1, const float* __restrict__ Wm2,
                           const float* __restrict__ bm2, float* __restrict__ out) {
    __shared__ __bf16 sAh[16 * EK_STRIDE];
    __shared__ __bf16 sAl[16 * EK_STRIDE];
    __shared__ float  part[16];
    __shared__ int    sSrc[16], sDst[16];

    const int tile = blockIdx.x;
    if (threadIdx.x < 16) {
        int e = tile * 16 + threadIdx.x;
        sSrc[threadIdx.x] = src[e];
        sDst[threadIdx.x] = dst[e];
        part[threadIdx.x] = 0.0f;
    }
    __syncthreads();

    // cooperative gather: 16 rows x 48 chunks of 8 bf16 (16B vector copies)
    {
        uint4* dh = (uint4*)sAh;
        uint4* dl = (uint4*)sAl;
        const uint4* gh = (const uint4*)xcat_h;
        const uint4* gl = (const uint4*)xcat_l;
        for (int t = threadIdx.x; t < 16 * (EDGE_K / 8); t += blockDim.x) {
            int r  = t / (EDGE_K / 8);
            int ch = t - r * (EDGE_K / 8);     // chunk of 8 elems
            int k  = ch * 8;
            int nd = (k < X_CAT_C) ? sSrc[r] : sDst[r];
            int kk = (k < X_CAT_C) ? k : k - X_CAT_C;
            size_t gi = (size_t)nd * (X_CAT_C / 8) + (kk >> 3);
            int    di = r * (EK_STRIDE / 8) + ch;
            dh[di] = gh[gi];
            dl[di] = gl[gi];
        }
    }
    __syncthreads();

    const int lane = threadIdx.x & 31;
    const int wave = threadIdx.x >> 5;
    const int m    = lane & 15;
    const int hh   = lane >> 4;
    const int ocol = wave * 16 + m;
    const __bf16* ah_row = sAh + m * EK_STRIDE;
    const __bf16* al_row = sAl + m * EK_STRIDE;
    const __bf16* bh_row = Wh  + (size_t)ocol * EDGE_K;
    const __bf16* bl_row = Wlo + (size_t)ocol * EDGE_K;

    v8f c = {};
    #pragma unroll
    for (int kb = 0; kb < EDGE_K; kb += 32) {
        v16bf a_hi = load_afrag(ah_row, kb, hh);
        v16bf a_lo = load_afrag(al_row, kb, hh);
        v16bf b_hi = load_bfrag(bh_row, kb, hh);
        v16bf b_lo = load_bfrag(bl_row, kb, hh);
        c = __builtin_amdgcn_wmma_f32_16x16x32_bf16(false, a_hi, false, b_hi, (short)0, c, false, false);
        c = __builtin_amdgcn_wmma_f32_16x16x32_bf16(false, a_hi, false, b_lo, (short)0, c, false, false);
        c = __builtin_amdgcn_wmma_f32_16x16x32_bf16(false, a_lo, false, b_hi, (short)0, c, false, false);
    }

    // fuse bias + relu + second linear (OUT_C = 1)
    const float bias = bm1[ocol];
    const float w2   = Wm2[ocol];
    #pragma unroll
    for (int r = 0; r < 8; ++r) {
        float h = fmaxf(c[r] + bias, 0.0f);
        atomicAdd(&part[r + hh * 8], h * w2);
    }
    __syncthreads();
    if (threadIdx.x < 16)
        out[tile * 16 + threadIdx.x] = part[threadIdx.x] + bm2[0];
}

// ---------------------------------------------------------------------------
extern "C" void kernel_launch(void* const* d_in, const int* in_sizes, int n_in,
                              void* d_out, int out_size, void* d_ws, size_t ws_size,
                              hipStream_t stream) {
    (void)in_sizes; (void)n_in; (void)out_size; (void)ws_size;
    const float* x   = (const float*)d_in[0];
    const int*   src = (const int*)  d_in[1];
    const int*   dst = (const int*)  d_in[2];
    const float* Wl[3] = {(const float*)d_in[3], (const float*)d_in[6], (const float*)d_in[9]};
    const float* bl[3] = {(const float*)d_in[4], (const float*)d_in[7], (const float*)d_in[10]};
    const float* Wr[3] = {(const float*)d_in[5], (const float*)d_in[8], (const float*)d_in[11]};
    const float* Wm1 = (const float*)d_in[12];
    const float* bm1 = (const float*)d_in[13];
    const float* Wm2 = (const float*)d_in[14];
    const float* bm2 = (const float*)d_in[15];
    float* out = (float*)d_out;

    // workspace layout (~154 MB total)
    float*  deg    = (float*)d_ws;
    float*  msg    = deg + N_NODES;
    float*  xcat   = msg + (size_t)N_NODES * IN_C;
    __bf16* xcat_h = (__bf16*)(xcat + (size_t)N_NODES * X_CAT_C);
    __bf16* xcat_l = xcat_h + (size_t)N_NODES * X_CAT_C;
    __bf16* Ah     = xcat_l + (size_t)N_NODES * X_CAT_C;
    __bf16* Al     = Ah + (size_t)N_NODES * 2 * IN_C;
    __bf16* Wch    = Al + (size_t)N_NODES * 2 * IN_C;
    __bf16* Wcl    = Wch + 64 * 2 * IN_C;
    __bf16* Wm1h   = Wcl + 64 * 2 * IN_C;
    __bf16* Wm1l   = Wm1h + 64 * EDGE_K;

    hipMemsetAsync(deg, 0, (size_t)N_NODES * sizeof(float), stream);
    k_degree<<<(N_EDGES + 255) / 256, 256, 0, stream>>>(dst, deg);
    k_prep_wm1<<<64, EDGE_K, 0, stream>>>(Wm1, Wm1h, Wm1l);

    const float* hprev = x;
    int hstride = IN_C;
    for (int l = 0; l < 3; ++l) {
        int fin = (l == 0) ? IN_C : HID;
        hipMemsetAsync(msg, 0, (size_t)N_NODES * fin * sizeof(float), stream);
        k_scatter<<<N_EDGES, fin, 0, stream>>>(hprev, hstride, fin, src, dst, msg);
        k_prep_node_A<<<N_NODES, 2 * fin, 0, stream>>>(msg, deg, hprev, hstride, fin, Ah, Al);
        k_prep_wcat<<<64, 2 * fin, 0, stream>>>(Wl[l], Wr[l], fin, Wch, Wcl);
        if (l == 0)
            k_sage_gemm<2 * IN_C><<<N_NODES / 16, 128, 0, stream>>>(
                Ah, Al, Wch, Wcl, bl[l], xcat + l * HID, xcat_h + l * HID, xcat_l + l * HID);
        else
            k_sage_gemm<2 * HID><<<N_NODES / 16, 128, 0, stream>>>(
                Ah, Al, Wch, Wcl, bl[l], xcat + l * HID, xcat_h + l * HID, xcat_l + l * HID);
        hprev   = xcat + l * HID;
        hstride = X_CAT_C;
    }

    k_edge_mlp<<<N_EDGES / 16, 128, 0, stream>>>(xcat_h, xcat_l, src, dst,
                                                 Wm1h, Wm1l, bm1, Wm2, bm2, out);
}
